// FraudGAT_19945828123269
// MI455X (gfx1250) — matrix operand order repaired and checked by reading, not compiled, not run on hardware
//
#include <hip/hip_runtime.h>
#include <hip/hip_bf16.h>

// ---------------------------------------------------------------------------
// FraudGAT on MI455X (gfx1250, wave32).
// GEMMs: V_WMMA_F32_16X16X32_BF16 with the B operand staged into LDS once per
// block via the Tensor Data Mover (tensor_load_to_lds + s_wait_tensorcnt),
// giving 8x B-operand reuse across the block's 8 waves.
// Edge softmax/aggregation uses L2 float atomics (memory-bound phase).
// ---------------------------------------------------------------------------

typedef __bf16 bf16_t;
typedef bf16_t v16bf __attribute__((ext_vector_type(16)));
typedef float  v8f   __attribute__((ext_vector_type(8)));
typedef unsigned int v4u __attribute__((ext_vector_type(4)));
typedef int v4i __attribute__((ext_vector_type(4)));
typedef int v8i __attribute__((ext_vector_type(8)));

#if defined(__HIP_DEVICE_COMPILE__) && defined(__has_builtin)
#  if __has_builtin(__builtin_amdgcn_tensor_load_to_lds)
#    define USE_TDM 1
#  else
#    define USE_TDM 0
#  endif
#else
#  define USE_TDM 0
#endif

#if defined(__has_include)
#  if __has_include(<hip/amd_detail/amd_gfx1250_TDM.h>)
#    define TDM_6ARG 1
#  else
#    define TDM_6ARG 0
#  endif
#else
#  define TDM_6ARG 0
#endif

static __host__ __device__ inline int cdiv_i(int a, int b) { return (a + b - 1) / b; }

// ---- monotonic float<->uint encoding for atomic segment-max ---------------
__device__ __forceinline__ unsigned fenc(float x) {
    unsigned u = __float_as_uint(x);
    return (u & 0x80000000u) ? ~u : (u | 0x80000000u);
}
__device__ __forceinline__ float fdec(unsigned k) {
    return (k & 0x80000000u) ? __uint_as_float(k & 0x7fffffffu) : __uint_as_float(~k);
}

// ---------------------------------------------------------------------------
// Layer-1 GEMM: [N,5] x [5,512] -> [N,512]. K=5 -> VALU kernel, float4 stores.
// ---------------------------------------------------------------------------
__global__ __launch_bounds__(256) void gemm_in5_kernel(
    const float* __restrict__ X, const float* __restrict__ W,
    float* __restrict__ Out, int Nn) {
    int t = blockIdx.x * blockDim.x + threadIdx.x;
    if (t >= Nn * 128) return;            // 128 float4's of output per node
    int n = t >> 7;
    int j = (t & 127) << 2;
    float xr[5];
#pragma unroll
    for (int k = 0; k < 5; ++k) xr[k] = X[(size_t)n * 5 + k];
    float4 acc = make_float4(0.f, 0.f, 0.f, 0.f);
#pragma unroll
    for (int k = 0; k < 5; ++k) {
        float4 wv = *(const float4*)(W + (size_t)k * 512 + j);
        acc.x += xr[k] * wv.x; acc.y += xr[k] * wv.y;
        acc.z += xr[k] * wv.z; acc.w += xr[k] * wv.w;
    }
    *(float4*)(Out + (size_t)n * 512 + j) = acc;
}

// ---------------------------------------------------------------------------
// Pack W[K,N] (f32 row-major) into WMMA B-operand fragments (bf16).
// Layout: [K/32][N/16][lane 0..31][16 bf16], where for tile (kt,nt):
//   col  = nt*16 + (lane&15)
//   krow = kt*32 + (lane>>4)*16     (lanes 0-15: K 0..15, lanes 16-31: K 16..31)
// ---------------------------------------------------------------------------
__global__ __launch_bounds__(256) void pack_w_kernel(
    const float* __restrict__ W, bf16_t* __restrict__ Bp, int K, int N) {
    int t = blockIdx.x * blockDim.x + threadIdx.x;
    int KT = K >> 5, NT = N >> 4;
    if (t >= KT * NT * 32) return;
    int lane = t & 31;
    int nt = (t >> 5) % NT;
    int kt = (t >> 5) / NT;
    int col  = (nt << 4) + (lane & 15);
    int krow = (kt << 5) + ((lane >> 4) << 4);
    bf16_t* out = Bp + (size_t)t * 16;
#pragma unroll
    for (int i = 0; i < 16; ++i)
        out[i] = (bf16_t)W[(size_t)(krow + i) * N + col];
}

// ---------------------------------------------------------------------------
// WMMA GEMM, K fixed at 512. Block = 8 waves, all sharing one 64-wide N strip
// (ns = blockIdx.y); wave w handles M-tile mt = blockIdx.x*8 + w.
// The block's B strip (16 k-steps x 4 tiles x 1KB = 64KB) is DMA'd into LDS
// once by the Tensor Data Mover, then all 8 waves feed WMMA from ds_load.
// A is loaded from global as f32 and converted to bf16 in-register:
//   m = mt*16 + (lane&15); kgrp = lane>>4
//   elems 0..15 = A[m][kk + kgrp*16 + 0..15]
// ---------------------------------------------------------------------------
__global__ __launch_bounds__(256) void wmma_gemm_kernel(
    const float* __restrict__ A, const bf16_t* __restrict__ Bp,
    float* __restrict__ C, int M, int Ntiles) {
    extern __shared__ char ldsB[];                 // 64KB dynamic LDS
    const int lane = threadIdx.x & 31;
    const int wave = threadIdx.x >> 5;
    const int ns = blockIdx.y;                     // 64-wide N strip
    const int mt = blockIdx.x * 8 + wave;          // wave-uniform
    const bool active = (mt < (M >> 4));
    const size_t stripB = (size_t)ns * 4096;       // byte offset of strip row 0

#if USE_TDM
    if (threadIdx.x < 32) {                        // wave 0 issues the DMA
        unsigned ldsoff = (unsigned)(uintptr_t)(void*)ldsB;   // low 32b = LDS offset
        unsigned long long ga =
            (unsigned long long)(uintptr_t)((const char*)Bp + stripB);
        // D# group 0: count=1 | lds_addr | global_addr | type=2
        v4u g0 = { 1u,
                   ldsoff,
                   (unsigned)(ga & 0xffffffffull),
                   (unsigned)((ga >> 32) & 0x01ffffffull) | 0x80000000u };
        // D# group 1: 2-D tensor == tile: 512 x 16 elements of 8B,
        // row stride = Ntiles*128 elements (the full packed-B row).
        unsigned stride0 = (unsigned)(Ntiles * 128);
        v8i g1 = { (int)0x00030000,        // wg_mask=0, data_size=3 (8B)
                   (int)(512u << 16),      // tensor_dim0[15:0]=512
                   (int)(16u << 16),       // tensor_dim0 hi=0 | tensor_dim1[15:0]=16
                   (int)(512u << 16),      // tensor_dim1 hi=0 | tile_dim0=512
                   (int)16,                // tile_dim1=16, tile_dim2=0
                   (int)stride0,           // tensor_dim0_stride[31:0]
                   0, 0 };                 // stride0 hi=0, dim1_stride=0
        v4i gz = { 0, 0, 0, 0 };
#if TDM_6ARG
        v8i gz8 = { 0, 0, 0, 0, 0, 0, 0, 0 };
        __builtin_amdgcn_tensor_load_to_lds(g0, g1, gz, gz, gz8, 0);
#else
        __builtin_amdgcn_tensor_load_to_lds(g0, g1, gz, gz, 0);
#endif
        __builtin_amdgcn_s_wait_tensorcnt(0);
    }
#else
    // fallback: cooperative staging (16B per lane per step, 64KB total)
    for (int i = threadIdx.x; i < 4096; i += 256) {
        int kt = i >> 8, wi = i & 255;
        *(uint4*)(ldsB + (size_t)i * 16) =
            *(const uint4*)((const char*)Bp + stripB +
                            (size_t)kt * ((size_t)Ntiles * 1024) + (size_t)wi * 16);
    }
#endif
    __syncthreads();
    if (!active) return;

    const int kgrp = lane >> 4;
    const int ln = lane & 15;
    const float* arow = A + (size_t)((mt << 4) + ln) * 512;
    const int ldc = Ntiles << 4;

    v8f acc0 = {}; v8f acc1 = {}; v8f acc2 = {}; v8f acc3 = {};

    for (int kt = 0; kt < 16; ++kt) {
        const int kb = (kt << 5) + (kgrp << 4);
        float4 f0 = *(const float4*)(arow + kb);
        float4 f1 = *(const float4*)(arow + kb + 4);
        float4 g0 = *(const float4*)(arow + kb + 8);
        float4 g1 = *(const float4*)(arow + kb + 12);
        v16bf a;
        a[0] = (bf16_t)f0.x; a[1] = (bf16_t)f0.y; a[2]  = (bf16_t)f0.z; a[3]  = (bf16_t)f0.w;
        a[4] = (bf16_t)f1.x; a[5] = (bf16_t)f1.y; a[6]  = (bf16_t)f1.z; a[7]  = (bf16_t)f1.w;
        a[8] = (bf16_t)g0.x; a[9] = (bf16_t)g0.y; a[10] = (bf16_t)g0.z; a[11] = (bf16_t)g0.w;
        a[12]= (bf16_t)g1.x; a[13]= (bf16_t)g1.y; a[14] = (bf16_t)g1.z; a[15] = (bf16_t)g1.w;

        const char* lb = ldsB + ((size_t)kt << 12) + ((size_t)lane << 5);
        v16bf b0 = *(const v16bf*)(lb);
        v16bf b1 = *(const v16bf*)(lb + 1024);
        v16bf b2 = *(const v16bf*)(lb + 2048);
        v16bf b3 = *(const v16bf*)(lb + 3072);

        acc0 = __builtin_amdgcn_wmma_f32_16x16x32_bf16(false, a, false, b0, (short)0, acc0, false, false);
        acc1 = __builtin_amdgcn_wmma_f32_16x16x32_bf16(false, a, false, b1, (short)0, acc1, false, false);
        acc2 = __builtin_amdgcn_wmma_f32_16x16x32_bf16(false, a, false, b2, (short)0, acc2, false, false);
        acc3 = __builtin_amdgcn_wmma_f32_16x16x32_bf16(false, a, false, b3, (short)0, acc3, false, false);
    }

    // C/D layout: VGPR r -> row = mt*16 + (lane>>4)*8 + r, col = (lane&15) in tile
    const int row0 = (mt << 4) + (kgrp << 3);
    const int colb = (ns << 6) + ln;
#pragma unroll
    for (int r = 0; r < 8; ++r) {
        float* crow = C + (size_t)(row0 + r) * ldc + colb;
        crow[0]  = acc0[r];
        crow[16] = acc1[r];
        crow[32] = acc2[r];
        crow[48] = acc3[r];
    }
}

// ---------------------------------------------------------------------------
// Per-(node,head) attention score dots: s_src = <h, a_src>, s_dst = <h, a_dst>
// ---------------------------------------------------------------------------
template <int H>
__global__ __launch_bounds__(256) void scores_kernel(
    const float* __restrict__ Hlin, const float* __restrict__ Asrc,
    const float* __restrict__ Adst, float* __restrict__ Ssrc,
    float* __restrict__ Sdst, int Nn) {
    int t = blockIdx.x * blockDim.x + threadIdx.x;
    if (t >= Nn * H) return;
    int h = t % H;
    const float4* hp = (const float4*)(Hlin + (size_t)t * 64);
    const float4* ap = (const float4*)(Asrc + (size_t)h * 64);
    const float4* dp = (const float4*)(Adst + (size_t)h * 64);
    float ss = 0.f, sd = 0.f;
#pragma unroll
    for (int i = 0; i < 16; ++i) {
        float4 hv = hp[i], av = ap[i], bv = dp[i];
        ss += hv.x * av.x + hv.y * av.y + hv.z * av.z + hv.w * av.w;
        sd += hv.x * bv.x + hv.y * bv.y + hv.z * bv.z + hv.w * bv.w;
    }
    Ssrc[t] = ss;
    Sdst[t] = sd;
}

__device__ __forceinline__ void edge_ends(const int* ei, int E, int e, int& s, int& d) {
    if (e < E) { s = ei[e]; d = ei[E + e]; }
    else       { s = e - E; d = s; }        // appended self-loops
}

// pass 1: segment max of leaky_relu(s_src[src]+s_dst[dst]) over dst
template <int H>
__global__ __launch_bounds__(256) void edge_max_kernel(
    const int* __restrict__ ei, int E, int Etot,
    const float* __restrict__ Ssrc, const float* __restrict__ Sdst,
    unsigned* __restrict__ Menc) {
    int e = blockIdx.x * blockDim.x + threadIdx.x;
    if (e >= Etot) return;
    int s, d; edge_ends(ei, E, e, s, d);
#pragma unroll
    for (int h = 0; h < H; ++h) {
        float v = Ssrc[(size_t)s * H + h] + Sdst[(size_t)d * H + h];
        v = v > 0.f ? v : 0.2f * v;
        atomicMax(Menc + (size_t)d * H + h, fenc(v));
    }
}

// pass 2: p = exp(e - m[dst]); store p; denom[dst] += p
template <int H>
__global__ __launch_bounds__(256) void edge_sum_kernel(
    const int* __restrict__ ei, int E, int Etot,
    const float* __restrict__ Ssrc, const float* __restrict__ Sdst,
    const unsigned* __restrict__ Menc, float* __restrict__ Denom,
    float* __restrict__ Pbuf) {
    int e = blockIdx.x * blockDim.x + threadIdx.x;
    if (e >= Etot) return;
    int s, d; edge_ends(ei, E, e, s, d);
#pragma unroll
    for (int h = 0; h < H; ++h) {
        float v = Ssrc[(size_t)s * H + h] + Sdst[(size_t)d * H + h];
        v = v > 0.f ? v : 0.2f * v;
        float m = fdec(Menc[(size_t)d * H + h]);
        if (!__builtin_isfinite(m)) m = 0.f;   // matches reference guard
        float p = expf(v - m);
        Pbuf[(size_t)e * H + h] = p;
        atomicAdd(Denom + (size_t)d * H + h, p);
    }
}

// pass 3: agg[dst,h,:] += alpha * h[src,h,:]
template <int H>
__global__ __launch_bounds__(256) void edge_agg_kernel(
    const int* __restrict__ ei, int E, int Etot,
    const float* __restrict__ Hlin, const float* __restrict__ Pbuf,
    const float* __restrict__ Denom, float* __restrict__ Agg) {
    long long t = (long long)blockIdx.x * blockDim.x + threadIdx.x;
    if (t >= (long long)Etot * H) return;
    int e = (int)(t / H);
    int h = (int)(t - (long long)e * H);
    int s, d; edge_ends(ei, E, e, s, d);
    float alpha = Pbuf[t] / (Denom[(size_t)d * H + h] + 1e-16f);
    const float4* hs = (const float4*)(Hlin + ((size_t)s * H + h) * 64);
    float* ag = Agg + ((size_t)d * H + h) * 64;
#pragma unroll
    for (int i = 0; i < 16; ++i) {
        float4 v = hs[i];
        atomicAdd(ag + 4 * i + 0, alpha * v.x);
        atomicAdd(ag + 4 * i + 1, alpha * v.y);
        atomicAdd(ag + 4 * i + 2, alpha * v.z);
        atomicAdd(ag + 4 * i + 3, alpha * v.w);
    }
}

// bias + ELU, in place
__global__ __launch_bounds__(256) void bias_elu_kernel(
    float* __restrict__ Buf, const float* __restrict__ B, int total, int HC) {
    int t = blockIdx.x * blockDim.x + threadIdx.x;
    if (t >= total) return;
    float v = Buf[t] + B[t % HC];
    Buf[t] = v > 0.f ? v : expm1f(v);
}

// classifier: sigmoid(h3 @ Wc + bc)
__global__ __launch_bounds__(256) void classifier_kernel(
    const float* __restrict__ F, const float* __restrict__ Wc,
    const float* __restrict__ bc, float* __restrict__ Out, int Nn) {
    int n = blockIdx.x * blockDim.x + threadIdx.x;
    if (n >= Nn) return;
    const float4* f = (const float4*)(F + (size_t)n * 64);
    const float4* w = (const float4*)Wc;
    float s = 0.f;
#pragma unroll
    for (int i = 0; i < 16; ++i) {
        float4 a = f[i], b = w[i];
        s += a.x * b.x + a.y * b.y + a.z * b.z + a.w * b.w;
    }
    s += bc[0];
    Out[n] = 1.f / (1.f + expf(-s));
}

// ---------------------------------------------------------------------------
extern "C" void kernel_launch(void* const* d_in, const int* in_sizes, int n_in,
                              void* d_out, int out_size, void* d_ws, size_t ws_size,
                              hipStream_t stream) {
    (void)n_in; (void)out_size; (void)ws_size;
    const float* x   = (const float*)d_in[0];
    const int*   ei  = (const int*)d_in[1];
    const float* W1  = (const float*)d_in[2];
    const float* as1 = (const float*)d_in[3];
    const float* ad1 = (const float*)d_in[4];
    const float* b1  = (const float*)d_in[5];
    const float* W2  = (const float*)d_in[6];
    const float* as2 = (const float*)d_in[7];
    const float* ad2 = (const float*)d_in[8];
    const float* b2  = (const float*)d_in[9];
    const float* W3  = (const float*)d_in[10];
    const float* as3 = (const float*)d_in[11];
    const float* ad3 = (const float*)d_in[12];
    const float* b3  = (const float*)d_in[13];
    const float* Wc  = (const float*)d_in[14];
    const float* bc  = (const float*)d_in[15];
    float* out = (float*)d_out;

    const int Nn   = in_sizes[0] / 5;   // 50000
    const int E    = in_sizes[1] / 2;   // 400000
    const int Etot = E + Nn;            // + self loops
    const int MT   = Nn >> 4;           // 3125 M-tiles

    // workspace carve-up (~223 MB; all chunks keep 32B alignment)
    char* w = (char*)d_ws;
    float* Bf0 = (float*)w;                 w += (size_t)Nn * 512 * sizeof(float); // features / agg
    float* Bf1 = (float*)w;                 w += (size_t)Nn * 512 * sizeof(float); // hlin (x@W)
    float* ssrc = (float*)w;                w += (size_t)Nn * 8 * sizeof(float);
    float* sdst = (float*)w;                w += (size_t)Nn * 8 * sizeof(float);
    unsigned* menc = (unsigned*)w;          w += (size_t)Nn * 8 * sizeof(unsigned);
    float* denom = (float*)w;               w += (size_t)Nn * 8 * sizeof(float);
    float* pbuf = (float*)w;                w += (size_t)Etot * 8 * sizeof(float);
    bf16_t* wpack = (bf16_t*)w;             w += (size_t)512 * 512 * sizeof(bf16_t);

    const int T = 256;
    const size_t LDSB = 64 * 1024;          // dynamic LDS for B staging

    // ---------------- Layer 1: in=5 -> 8x64 ----------------
    gemm_in5_kernel<<<cdiv_i(Nn * 128, T), T, 0, stream>>>(x, W1, Bf1, Nn);
    scores_kernel<8><<<cdiv_i(Nn * 8, T), T, 0, stream>>>(Bf1, as1, ad1, ssrc, sdst, Nn);
    hipMemsetAsync(menc, 0, (size_t)Nn * 8 * 4, stream);
    hipMemsetAsync(denom, 0, (size_t)Nn * 8 * 4, stream);
    hipMemsetAsync(Bf0, 0, (size_t)Nn * 512 * 4, stream);
    edge_max_kernel<8><<<cdiv_i(Etot, T), T, 0, stream>>>(ei, E, Etot, ssrc, sdst, menc);
    edge_sum_kernel<8><<<cdiv_i(Etot, T), T, 0, stream>>>(ei, E, Etot, ssrc, sdst, menc, denom, pbuf);
    edge_agg_kernel<8><<<cdiv_i(Etot * 8, T), T, 0, stream>>>(ei, E, Etot, Bf1, pbuf, denom, Bf0);
    bias_elu_kernel<<<cdiv_i(Nn * 512, T), T, 0, stream>>>(Bf0, b1, Nn * 512, 512);

    // ---------------- Layer 2: 512 -> 8x64 (WMMA GEMM, TDM-staged B) -------
    pack_w_kernel<<<cdiv_i(16 * 32 * 32, T), T, 0, stream>>>(W2, wpack, 512, 512);
    wmma_gemm_kernel<<<dim3(cdiv_i(MT, 8), 8), T, LDSB, stream>>>(Bf0, wpack, Bf1, Nn, 32);
    scores_kernel<8><<<cdiv_i(Nn * 8, T), T, 0, stream>>>(Bf1, as2, ad2, ssrc, sdst, Nn);
    hipMemsetAsync(menc, 0, (size_t)Nn * 8 * 4, stream);
    hipMemsetAsync(denom, 0, (size_t)Nn * 8 * 4, stream);
    hipMemsetAsync(Bf0, 0, (size_t)Nn * 512 * 4, stream);
    edge_max_kernel<8><<<cdiv_i(Etot, T), T, 0, stream>>>(ei, E, Etot, ssrc, sdst, menc);
    edge_sum_kernel<8><<<cdiv_i(Etot, T), T, 0, stream>>>(ei, E, Etot, ssrc, sdst, menc, denom, pbuf);
    edge_agg_kernel<8><<<cdiv_i(Etot * 8, T), T, 0, stream>>>(ei, E, Etot, Bf1, pbuf, denom, Bf0);
    bias_elu_kernel<<<cdiv_i(Nn * 512, T), T, 0, stream>>>(Bf0, b2, Nn * 512, 512);

    // ---------------- Layer 3: 512 -> 1x64 (WMMA GEMM, TDM-staged B) -------
    pack_w_kernel<<<cdiv_i(16 * 4 * 32, T), T, 0, stream>>>(W3, wpack, 512, 64);
    wmma_gemm_kernel<<<dim3(cdiv_i(MT, 8), 1), T, LDSB, stream>>>(Bf0, wpack, Bf1, Nn, 4);
    scores_kernel<1><<<cdiv_i(Nn, T), T, 0, stream>>>(Bf1, as3, ad3, ssrc, sdst, Nn);
    hipMemsetAsync(menc, 0, (size_t)Nn * 4, stream);
    hipMemsetAsync(denom, 0, (size_t)Nn * 4, stream);
    hipMemsetAsync(Bf0, 0, (size_t)Nn * 64 * 4, stream);
    edge_max_kernel<1><<<cdiv_i(Etot, T), T, 0, stream>>>(ei, E, Etot, ssrc, sdst, menc);
    edge_sum_kernel<1><<<cdiv_i(Etot, T), T, 0, stream>>>(ei, E, Etot, ssrc, sdst, menc, denom, pbuf);
    edge_agg_kernel<1><<<cdiv_i(Etot, T), T, 0, stream>>>(ei, E, Etot, Bf1, pbuf, denom, Bf0);
    bias_elu_kernel<<<cdiv_i(Nn * 64, T), T, 0, stream>>>(Bf0, b3, Nn * 64, 64);

    // ---------------- Classifier ----------------
    classifier_kernel<<<cdiv_i(Nn, T), T, 0, stream>>>(Bf0, Wc, bc, out, Nn);
}
